// Outer_4440996184167
// MI455X (gfx1250) — matrix operand order repaired and checked by reading, not compiled, not run on hardware
//
#include <hip/hip_runtime.h>

#define BATCH 2048
#define NH    512      // hidden / output width
#define IPV   257      // number of i values (I1+1)
#define JPAD  288      // j padded to 3*96
#define APAD  260      // aT leading dim (i padded)
#define KPRIME (IPV * JPAD)          // 74016 padded K per n-row of W1t
#define W1T_ROW_UINTS (KPRIME / 2)   // 37008

typedef __attribute__((ext_vector_type(16))) __bf16        v16bf;
typedef __attribute__((ext_vector_type(8)))  float         v8f;
typedef __attribute__((ext_vector_type(8)))  unsigned int  v8u;

union Frag {
  v8u          v;
  unsigned int u[8];
  uint4        q[2];
};

__device__ __forceinline__ unsigned short f2bf_bits(float x) {
  unsigned int u = __builtin_bit_cast(unsigned int, x);
  u += 0x7FFFu + ((u >> 16) & 1u);            // round-to-nearest-even
  return (unsigned short)(u >> 16);
}
__device__ __forceinline__ unsigned int pack_bf16(float lo, float hi) {
  return (unsigned int)f2bf_bits(lo) | ((unsigned int)f2bf_bits(hi) << 16);
}
__device__ __forceinline__ unsigned int pk_mul_bf16(unsigned int a, unsigned int b) {
  unsigned int d;
  asm("v_pk_mul_bf16 %0, %1, %2" : "=v"(d) : "v"(a), "v"(b));
  return d;
}
__device__ __forceinline__ v8f wmma_bf16(const Frag& a, const Frag& b, v8f c) {
  return __builtin_amdgcn_wmma_f32_16x16x32_bf16(
      false, __builtin_bit_cast(v16bf, a.v),
      false, __builtin_bit_cast(v16bf, b.v),
      (short)0, c, false, false);
}

// ---------------- prep kernels ----------------
__global__ void prep_aT_kernel(const float* __restrict__ inp1, float* __restrict__ aT) {
  int idx = blockIdx.x * 256 + threadIdx.x;
  if (idx >= APAD * BATCH) return;
  int i = idx / BATCH, b = idx - i * BATCH;
  float v = 0.f;
  if (i < 256) v = inp1[b * 256 + i];
  else if (i == 256) v = 1.f;
  aT[idx] = v;   // aT[i][b]
}

__global__ void prep_c_kernel(const float* __restrict__ inp2, float* __restrict__ cF) {
  int idx = blockIdx.x * 256 + threadIdx.x;
  if (idx >= BATCH * JPAD) return;
  int b = idx / JPAD, j = idx - b * JPAD;
  float v = 0.f;
  if (j < 256) v = inp2[b * 256 + j];
  else if (j == 256) v = 1.f;
  cF[idx] = v;   // cF[b][j], zero-padded j>=257
}

__global__ void prep_w2_kernel(const float* __restrict__ W2, unsigned short* __restrict__ w2b) {
  int idx = blockIdx.x * 256 + threadIdx.x;
  if (idx >= NH * NH) return;
  w2b[idx] = f2bf_bits(W2[idx]);
}

// Transpose+pad+convert W1 [66049][512] f32  ->  W1t [512][74016] bf16.
// W1t[n][i*288 + j] = bf16(W1[i*257 + j][n]) for j<257, else 0.
__global__ __launch_bounds__(256) void prep_w1t_kernel(
    const float* __restrict__ W1, unsigned short* __restrict__ W1t) {
  __shared__ unsigned short tile[32][72];
  const int t = threadIdx.x;
  const int k0 = blockIdx.x * 32;   // base in padded-K space
  const int n0 = blockIdx.y * 64;
  #pragma unroll
  for (int p = 0; p < 8; ++p) {
    int idx = p * 256 + t;
    int kk = idx >> 6, nn = idx & 63;
    int kp = k0 + kk;
    int i = kp / JPAD;
    int j = kp - i * JPAD;
    float v = 0.f;
    if (j < 257) v = W1[(size_t)(i * 257 + j) * NH + n0 + nn];  // coalesced in nn
    tile[kk][nn] = f2bf_bits(v);
  }
  __syncthreads();
  #pragma unroll
  for (int p = 0; p < 4; ++p) {
    int idx = p * 256 + t;                   // 1024 dword writes
    int nn = idx >> 4, kk = (idx & 15) * 2;
    unsigned int v = (unsigned int)tile[kk][nn] |
                     ((unsigned int)tile[kk + 1][nn] << 16);
    *(unsigned int*)(W1t + (size_t)(n0 + nn) * KPRIME + k0 + kk) = v;  // coalesced in kk
  }
}

// ---------------- GEMM 1: h = relu(fusion @ W1 + b1) ----------------
// WG tile 128(M) x 128(N), 8 waves (4 along M x 2 along N), wave tile 32x64.
// K-step = 96 (3 x 32-K WMMA sub-steps per barrier pair).
#define CS_STRIDE 148                    // dwords per c-row (144 data + 4 pad)
#define CS_UINTS  (128 * CS_STRIDE)      // 75,776 B
#define BS_STRIDE 52                     // dwords per Bs row (48 data + 4 pad)
#define BS_ROWS   128
#define BS_UINTS  (BS_ROWS * BS_STRIDE)  // 26,624 B

__global__ __launch_bounds__(256) void gemm1_kernel(
    const float* __restrict__ aT, const float* __restrict__ cF,
    const unsigned int* __restrict__ W1t,   // bf16 pairs, [512][37008 dwords]
    const float* __restrict__ b1,
    unsigned short* __restrict__ hB) {
  extern __shared__ unsigned int smem[];
  unsigned int* cs = smem;             // [128][CS_STRIDE] bf16 pairs of c rows
  unsigned int* bs = smem + CS_UINTS;  // [128][BS_STRIDE] W1 tile, [n][k]

  const int t      = threadIdx.x;
  const int lane   = t & 31;
  const int laneLo = lane & 15;
  const int hiHalf = lane >> 4;
  const int wave   = t >> 5;
  const int waveM  = wave & 3;
  const int waveN  = wave >> 2;
  const int m0 = blockIdx.x * 128;
  const int n0 = blockIdx.y * 128;

  // Stage this WG's 128 c-rows into LDS as bf16 (once).
  {
    const int row = t >> 1, half = t & 1;
    const float2* src = (const float2*)(cF + (size_t)(m0 + row) * JPAD) + half * 72;
    unsigned int* dst = cs + row * CS_STRIDE + half * 72;
    #pragma unroll 4
    for (int q = 0; q < 72; ++q) {
      float2 x = src[q];
      dst[q] = pack_bf16(x.x, x.y);
    }
  }

  v8f acc[2][4];
  #pragma unroll
  for (int a = 0; a < 2; ++a)
    #pragma unroll
    for (int b = 0; b < 4; ++b)
      acc[a][b] = v8f{0.f, 0.f, 0.f, 0.f, 0.f, 0.f, 0.f, 0.f};

  // B-fill mapping: thread owns (column bn, 16-bf16 half bh of each 32-K chunk).
  const int bn = t & 127;
  const int bh = (t >> 7) * 8;  // dword offset 0 or 8 within a 16-dword chunk
  const unsigned int* w1row = W1t + (size_t)(n0 + bn) * W1T_ROW_UINTS + bh;
  unsigned int* bsdst = bs + bn * BS_STRIDE + bh;

  // software pipeline: preload (i=0, jb=0)
  uint4 st[6];
  #pragma unroll
  for (int kk = 0; kk < 3; ++kk) {
    st[kk * 2 + 0] = *(const uint4*)(w1row + kk * 16);
    st[kk * 2 + 1] = *(const uint4*)(w1row + kk * 16 + 4);
  }

  for (int i = 0; i < IPV; ++i) {
    // Broadcast a[b,i] per lane-row as a packed bf16 pair.
    unsigned int aPk[2];
    #pragma unroll
    for (int mt = 0; mt < 2; ++mt) {
      float av = aT[i * BATCH + m0 + waveM * 32 + mt * 16 + laneLo];
      unsigned int ab = f2bf_bits(av);
      aPk[mt] = ab | (ab << 16);
    }

    for (int jb = 0; jb < 3; ++jb) {       // 3 x 96-K macro steps per i
      __syncthreads();                     // previous step's consumers done
      #pragma unroll
      for (int kk = 0; kk < 3; ++kk) {     // commit staged W1 chunks
        *(uint4*)(bsdst + kk * 16)     = st[kk * 2 + 0];
        *(uint4*)(bsdst + kk * 16 + 4) = st[kk * 2 + 1];
      }
      __syncthreads();

      // prefetch next macro step's W1 chunks while WMMAs run
      {
        int ni = i, njb = jb + 1;
        if (njb == 3) { njb = 0; ++ni; }
        if (ni < IPV) {
          const unsigned int* src = w1row + ni * 144 + njb * 48;
          #pragma unroll
          for (int kk = 0; kk < 3; ++kk) {
            st[kk * 2 + 0] = *(const uint4*)(src + kk * 16);
            st[kk * 2 + 1] = *(const uint4*)(src + kk * 16 + 4);
          }
        }
      }

      #pragma unroll
      for (int kk = 0; kk < 3; ++kk) {     // 3 x 32-K WMMA sub-steps
        // B fragments: lane holds one column, 16 consecutive K (per half).
        Frag bfr[4];
        #pragma unroll
        for (int nt = 0; nt < 4; ++nt) {
          const uint4* p = (const uint4*)(bs + (waveN * 64 + nt * 16 + laneLo) * BS_STRIDE +
                                          kk * 16 + hiHalf * 8);
          bfr[nt].q[0] = p[0];
          bfr[nt].q[1] = p[1];
        }

        // A fragments built in registers: a[b,i] * c[b, jb*96 + kk*32 + k].
        #pragma unroll
        for (int mt = 0; mt < 2; ++mt) {
          const unsigned int* crow =
              cs + (waveM * 32 + mt * 16 + laneLo) * CS_STRIDE + jb * 48 + kk * 16;
          uint4 u0 = *(const uint4*)(crow + hiHalf * 8);       // K chunk {0|16 .. +8}
          uint4 u1 = *(const uint4*)(crow + hiHalf * 8 + 4);   // K chunk {8|24 .. +8}
          const unsigned int am = aPk[mt];
          Frag af;
          af.u[0] = pk_mul_bf16(u0.x, am);
          af.u[1] = pk_mul_bf16(u0.y, am);
          af.u[2] = pk_mul_bf16(u0.z, am);
          af.u[3] = pk_mul_bf16(u0.w, am);
          af.u[4] = pk_mul_bf16(u1.x, am);
          af.u[5] = pk_mul_bf16(u1.y, am);
          af.u[6] = pk_mul_bf16(u1.z, am);
          af.u[7] = pk_mul_bf16(u1.w, am);

          #pragma unroll
          for (int nt = 0; nt < 4; ++nt)
            acc[mt][nt] = wmma_bf16(af, bfr[nt], acc[mt][nt]);
        }
      }
    }
  }

  // Epilogue: + b1, relu, write bf16 h.
  #pragma unroll
  for (int nt = 0; nt < 4; ++nt) {
    const int col = n0 + waveN * 64 + nt * 16 + laneLo;
    const float bias = b1[col];
    #pragma unroll
    for (int mt = 0; mt < 2; ++mt) {
      const int rbase = m0 + waveM * 32 + mt * 16 + hiHalf * 8;
      #pragma unroll
      for (int v = 0; v < 8; ++v) {
        float x = acc[mt][nt][v] + bias;
        x = x > 0.f ? x : 0.f;
        hB[(size_t)(rbase + v) * NH + col] = f2bf_bits(x);
      }
    }
  }
}

// ---------------- GEMM 2: out = relu(h @ W2 + b2) ----------------
#define B2_STRIDE 20
__global__ __launch_bounds__(256) void gemm2_kernel(
    const unsigned short* __restrict__ hB, const unsigned short* __restrict__ w2b,
    const float* __restrict__ b2, float* __restrict__ out) {
  __shared__ unsigned int as2[128 * B2_STRIDE];   // [128][20] h tile  [row][k]
  __shared__ unsigned int bs2[64 * B2_STRIDE];    // [64][20]  W2 tile [n][k]

  const int t      = threadIdx.x;
  const int lane   = t & 31;
  const int laneLo = lane & 15;
  const int hiHalf = lane >> 4;
  const int wave   = t >> 5;
  const int waveM  = wave & 3;
  const int waveN  = wave >> 2;
  const int m0 = blockIdx.x * 128;
  const int n0 = blockIdx.y * 64;

  v8f acc[2][2];
  #pragma unroll
  for (int a = 0; a < 2; ++a)
    #pragma unroll
    for (int b = 0; b < 2; ++b)
      acc[a][b] = v8f{0.f, 0.f, 0.f, 0.f, 0.f, 0.f, 0.f, 0.f};

  const int bn  = t & 63;
  const int bkk = (t >> 6) << 3;

  for (int kb = 0; kb < NH; kb += 32) {
    __syncthreads();
    // A tile: 16 bf16 per half-row, straight copy from hB.
    {
      const int row = t >> 1, half = t & 1;
      const uint4* src = (const uint4*)(hB + (size_t)(m0 + row) * NH + kb + half * 16);
      uint4* dst = (uint4*)(as2 + row * B2_STRIDE + half * 8);
      dst[0] = src[0];
      dst[1] = src[1];
    }
    // B tile: transpose-gather bf16 W2 columns.
    {
      unsigned int pk[4];
      #pragma unroll
      for (int s = 0; s < 4; ++s) {
        int k0 = kb + bkk + 2 * s;
        pk[s] = (unsigned int)w2b[k0 * NH + n0 + bn] |
                ((unsigned int)w2b[(k0 + 1) * NH + n0 + bn] << 16);
      }
      *(uint4*)(bs2 + bn * B2_STRIDE + (bkk >> 1)) = make_uint4(pk[0], pk[1], pk[2], pk[3]);
    }
    __syncthreads();

    Frag bfr[2];
    #pragma unroll
    for (int nt = 0; nt < 2; ++nt) {
      const uint4* p = (const uint4*)(bs2 + (waveN * 32 + nt * 16 + laneLo) * B2_STRIDE + hiHalf * 8);
      bfr[nt].q[0] = p[0];
      bfr[nt].q[1] = p[1];
    }
    #pragma unroll
    for (int mt = 0; mt < 2; ++mt) {
      const unsigned int* arow = as2 + (waveM * 32 + mt * 16 + laneLo) * B2_STRIDE;
      Frag af;
      af.q[0] = *(const uint4*)(arow + hiHalf * 8);
      af.q[1] = *(const uint4*)(arow + hiHalf * 8 + 4);
      acc[mt][0] = wmma_bf16(af, bfr[0], acc[mt][0]);
      acc[mt][1] = wmma_bf16(af, bfr[1], acc[mt][1]);
    }
  }

  #pragma unroll
  for (int nt = 0; nt < 2; ++nt) {
    const int col = n0 + waveN * 32 + nt * 16 + laneLo;
    const float bias = b2[col];
    #pragma unroll
    for (int mt = 0; mt < 2; ++mt) {
      const int rbase = m0 + waveM * 32 + mt * 16 + hiHalf * 8;
      #pragma unroll
      for (int v = 0; v < 8; ++v) {
        float x = acc[mt][nt][v] + bias;
        out[(size_t)(rbase + v) * NH + col] = x > 0.f ? x : 0.f;
      }
    }
  }
}

// ---------------- launcher ----------------
extern "C" void kernel_launch(void* const* d_in, const int* in_sizes, int n_in,
                              void* d_out, int out_size, void* d_ws, size_t ws_size,
                              hipStream_t stream) {
  (void)in_sizes; (void)n_in; (void)out_size; (void)ws_size;
  const float* inp1 = (const float*)d_in[0];
  const float* inp2 = (const float*)d_in[1];
  const float* W1   = (const float*)d_in[2];
  const float* b1   = (const float*)d_in[3];
  const float* W2   = (const float*)d_in[4];
  const float* b2   = (const float*)d_in[5];
  float* out = (float*)d_out;

  // workspace layout (256B aligned chunks)
  char* ws = (char*)d_ws;
  float*          aT  = (float*)ws;                                         // 260*2048*4   = 2,129,920
  float*          cF  = (float*)(ws + 2129920);                             // 2048*288*4   = 2,359,296
  unsigned short* w2b = (unsigned short*)(ws + 2129920 + 2359296);          // 512*512*2    =   524,288
  unsigned short* hB  = (unsigned short*)(ws + 2129920 + 2359296 + 524288); // 2048*512*2   = 2,097,152
  unsigned short* W1t = (unsigned short*)(ws + 7110656);                    // 512*74016*2  = 75,792,384
                                                                            // total ≈ 82.9 MB

  prep_aT_kernel<<<(APAD * BATCH + 255) / 256, 256, 0, stream>>>(inp1, aT);
  prep_c_kernel<<<(BATCH * JPAD + 255) / 256, 256, 0, stream>>>(inp2, cF);
  prep_w2_kernel<<<(NH * NH + 255) / 256, 256, 0, stream>>>(W2, w2b);

  dim3 gridT(KPRIME / 32, NH / 64);   // (2313, 8)
  prep_w1t_kernel<<<gridT, 256, 0, stream>>>(W1, W1t);

  dim3 grid1(BATCH / 128, NH / 128);  // (16, 4)
  size_t lds1 = (size_t)(CS_UINTS + BS_UINTS) * sizeof(unsigned int);  // 102,400 B
  gemm1_kernel<<<grid1, 256, lds1, stream>>>(aT, cF, (const unsigned int*)W1t, b1, hB);

  dim3 grid2(BATCH / 128, NH / 64);   // (16, 8)
  gemm2_kernel<<<grid2, 256, 0, stream>>>(hB, w2b, b2, out);
}
